// TransformerEncoderWithAttn_65996467470589
// MI455X (gfx1250) — matrix operand order repaired and checked by reading, not compiled, not run on hardware
//
#include <hip/hip_runtime.h>

// ---------------------------------------------------------------------------
// Types for CDNA5 (gfx1250) WMMA: V_WMMA_F32_16X16X32_BF16, wave32.
// ---------------------------------------------------------------------------
typedef __bf16 bf16;
typedef __attribute__((ext_vector_type(16))) __bf16 v16bf;
typedef __attribute__((ext_vector_type(8)))  __bf16 v8bf;
typedef __attribute__((ext_vector_type(8)))  float  v8f;

#define WPB 8  // waves per block (256 threads, wave32)

// ---------------------------------------------------------------------------
// Generic GEMM:  C[m,n] = alpha * sum_k A[m,k]*B[n,k]  (+bias[n]) (ReLU?)
// A: [M,K] bf16 row-major (lda), B: [N,K] bf16 row-major (ldb)  == A @ B^T
// Per-blockIdx.y/z pointer offsets allow batched attention (h = y, b = z).
// Each wave computes a 32(M) x 64(N) strip: 2 A-frags + 4 B-frags -> 8 WMMAs
// per 32-wide K step (every B fragment feeds two WMMAs -> 1.5 b128 loads per
// WMMA). Fragment loads are contiguous 128b global loads that give A and B
// the same contiguous k order, k = 16*half + s (not the ISA's table order).
// ---------------------------------------------------------------------------
__global__ void __launch_bounds__(256) gemm_bf16_nt(
    const bf16* __restrict__ A, int lda, long sAy, long sAz,
    const bf16* __restrict__ Bw, int ldb, long sBy, long sBz,
    float* __restrict__ Cf, bf16* __restrict__ Cb, int ldc, long sCy, long sCz,
    const float* __restrict__ bias,
    int M, int N, int K, float alpha, int relu)
{
    A  += (long)blockIdx.y * sAy + (long)blockIdx.z * sAz;
    Bw += (long)blockIdx.y * sBy + (long)blockIdx.z * sBz;
    const long coff = (long)blockIdx.y * sCy + (long)blockIdx.z * sCz;

    const int tilesN = N >> 6;          // 64-wide N strips
    const int tilesM = M >> 5;          // 32-tall M tiles
    const int wave = blockIdx.x * WPB + (threadIdx.x >> 5);
    if (wave >= tilesM * tilesN) return;        // wave-uniform: EXEC stays all-1s
    const int tm = wave / tilesN;
    const int tn = wave - tm * tilesN;

    const int lane = threadIdx.x & 31;
    const int half = lane >> 4;         // lane group 0..1
    const int ml   = lane & 15;

    // A-frags (16-bit 16x32): lane = row (m), halves hold K {0..15} / {16..31}
    const bf16* arow0 = A + (long)(tm * 32 + ml) * lda;
    const bf16* arow1 = arow0 + (long)16 * lda;
    // B-frag (16-bit 32x16): lane = col (n), 16 contiguous K values at half*16
    const bf16* brow = Bw + (long)(tn * 64 + ml) * ldb + half * 16;
    const long bstep = (long)16 * ldb;

    v8f acc00 = {}, acc01 = {}, acc02 = {}, acc03 = {};
    v8f acc10 = {}, acc11 = {}, acc12 = {}, acc13 = {};
    for (int k0 = 0; k0 < K; k0 += 32) {
        union { v16bf v; v8bf h[2]; } a0, a1;
        a0.h[0] = *(const v8bf*)(arow0 + k0 + half * 16);
        a0.h[1] = *(const v8bf*)(arow0 + k0 + 8 + half * 16);
        a1.h[0] = *(const v8bf*)(arow1 + k0 + half * 16);
        a1.h[1] = *(const v8bf*)(arow1 + k0 + 8 + half * 16);
        v16bf b0 = *(const v16bf*)(brow + k0);
        v16bf b1 = *(const v16bf*)(brow + bstep + k0);
        v16bf b2 = *(const v16bf*)(brow + 2 * bstep + k0);
        v16bf b3 = *(const v16bf*)(brow + 3 * bstep + k0);
        acc00 = __builtin_amdgcn_wmma_f32_16x16x32_bf16(false, a0.v, false, b0, (short)0, acc00, false, false);
        acc10 = __builtin_amdgcn_wmma_f32_16x16x32_bf16(false, a1.v, false, b0, (short)0, acc10, false, false);
        acc01 = __builtin_amdgcn_wmma_f32_16x16x32_bf16(false, a0.v, false, b1, (short)0, acc01, false, false);
        acc11 = __builtin_amdgcn_wmma_f32_16x16x32_bf16(false, a1.v, false, b1, (short)0, acc11, false, false);
        acc02 = __builtin_amdgcn_wmma_f32_16x16x32_bf16(false, a0.v, false, b2, (short)0, acc02, false, false);
        acc12 = __builtin_amdgcn_wmma_f32_16x16x32_bf16(false, a1.v, false, b2, (short)0, acc12, false, false);
        acc03 = __builtin_amdgcn_wmma_f32_16x16x32_bf16(false, a0.v, false, b3, (short)0, acc03, false, false);
        acc13 = __builtin_amdgcn_wmma_f32_16x16x32_bf16(false, a1.v, false, b3, (short)0, acc13, false, false);
    }

    // C/D layout: lane ml = N, VGPR r -> M = half*8 + r  (second frag at +16)
    const int m0 = tm * 32 + half * 8;
    v8f accs[2][4] = {{acc00, acc01, acc02, acc03}, {acc10, acc11, acc12, acc13}};
    #pragma unroll
    for (int c = 0; c < 4; ++c) {
        const int n = tn * 64 + c * 16 + ml;
        const float bv = bias ? bias[n] : 0.0f;
        #pragma unroll
        for (int g = 0; g < 2; ++g) {
            #pragma unroll
            for (int r = 0; r < 8; ++r) {
                float v = alpha * accs[g][c][r] + bv;
                if (relu) v = fmaxf(v, 0.0f);
                const long idx = coff + (long)(m0 + g * 16 + r) * ldc + n;
                if (Cf) Cf[idx] = v;
                if (Cb) Cb[idx] = (bf16)v;
            }
        }
    }
}

// ---------------------------------------------------------------------------
// fp32 -> bf16 elementwise convert (grid-stride)
// ---------------------------------------------------------------------------
__global__ void cvt_f32_bf16_k(const float* __restrict__ in, bf16* __restrict__ outp, long n)
{
    long i = (long)blockIdx.x * blockDim.x + threadIdx.x;
    const long stride = (long)gridDim.x * blockDim.x;
    for (; i < n; i += stride) outp[i] = (bf16)in[i];
}

// ---------------------------------------------------------------------------
// Build V^T: vT[b][h][d][j] = qkv_bf16[(b*512 + j)*1536 + 1024 + h*64 + d]
// so the context GEMM (attn @ V) becomes the same A@B^T form.
// ---------------------------------------------------------------------------
__global__ void build_vT_k(const bf16* __restrict__ qkv, bf16* __restrict__ vT)
{
    const long i = (long)blockIdx.x * 256 + threadIdx.x;   // 8*8*64*512 threads
    const int  j = (int)(i & 511);
    long t = i >> 9;
    const int d = (int)(t & 63); t >>= 6;
    const int h = (int)(t & 7);
    const int b = (int)(t >> 3);
    vT[i] = qkv[((long)(b * 512 + j)) * 1536 + 1024 + h * 64 + d];
}

// ---------------------------------------------------------------------------
// In-place softmax over 512-wide rows (one 256-thread block per row, wave32)
// ---------------------------------------------------------------------------
__global__ void __launch_bounds__(256) softmax512_k(float* __restrict__ p)
{
    float* rp = p + (long)blockIdx.x * 512;
    const int tid = threadIdx.x;
    float v0 = rp[tid], v1 = rp[tid + 256];

    float mx = fmaxf(v0, v1);
    #pragma unroll
    for (int o = 16; o; o >>= 1) mx = fmaxf(mx, __shfl_xor(mx, o, 32));
    __shared__ float redm[8];
    if ((tid & 31) == 0) redm[tid >> 5] = mx;
    __syncthreads();
    mx = redm[0];
    #pragma unroll
    for (int w = 1; w < 8; ++w) mx = fmaxf(mx, redm[w]);

    const float e0 = __expf(v0 - mx), e1 = __expf(v1 - mx);
    float s = e0 + e1;
    #pragma unroll
    for (int o = 16; o; o >>= 1) s += __shfl_xor(s, o, 32);
    __shared__ float reds[8];
    __syncthreads();
    if ((tid & 31) == 0) reds[tid >> 5] = s;
    __syncthreads();
    s = 0.0f;
    #pragma unroll
    for (int w = 0; w < 8; ++w) s += reds[w];

    const float inv = 1.0f / s;
    rp[tid] = e0 * inv;
    rp[tid + 256] = e1 * inv;
}

// ---------------------------------------------------------------------------
// Fused residual add + LayerNorm (512-wide), emits fp32 x and bf16 x.
// Safe for xin == xout (row read fully into registers first).
// ---------------------------------------------------------------------------
__global__ void __launch_bounds__(256) add_ln512_k(
    const float* __restrict__ xin, const float* __restrict__ delta,
    const float* __restrict__ gg, const float* __restrict__ bb,
    float* __restrict__ xout, bf16* __restrict__ xb)
{
    const long row = blockIdx.x;
    const int tid = threadIdx.x;
    const float* xr = xin + row * 512;
    const float* dr = delta + row * 512;
    const float v0 = xr[tid] + dr[tid];
    const float v1 = xr[tid + 256] + dr[tid + 256];

    float s = v0 + v1, ss = v0 * v0 + v1 * v1;
    #pragma unroll
    for (int o = 16; o; o >>= 1) { s += __shfl_xor(s, o, 32); ss += __shfl_xor(ss, o, 32); }
    __shared__ float rs[8], rss[8];
    if ((tid & 31) == 0) { rs[tid >> 5] = s; rss[tid >> 5] = ss; }
    __syncthreads();
    s = 0.0f; ss = 0.0f;
    #pragma unroll
    for (int w = 0; w < 8; ++w) { s += rs[w]; ss += rss[w]; }

    const float mu  = s * (1.0f / 512.0f);
    const float var = ss * (1.0f / 512.0f) - mu * mu;
    const float inv = rsqrtf(var + 1e-5f);
    const float o0 = (v0 - mu) * inv * gg[tid] + bb[tid];
    const float o1 = (v1 - mu) * inv * gg[tid + 256] + bb[tid + 256];
    xout[row * 512 + tid] = o0;
    xout[row * 512 + tid + 256] = o1;
    xb[row * 512 + tid] = (bf16)o0;
    xb[row * 512 + tid + 256] = (bf16)o1;
}

// ---------------------------------------------------------------------------
// Host side
// ---------------------------------------------------------------------------
static inline void gemm_launch(hipStream_t s,
    const bf16* A, int lda, long sAy, long sAz,
    const bf16* B, int ldb, long sBy, long sBz,
    float* Cf, bf16* Cb, int ldc, long sCy, long sCz,
    const float* bias, int M, int N, int K, float alpha, int relu, int gy, int gz)
{
    const int waves = (M / 32) * (N / 64);
    dim3 grid((waves + WPB - 1) / WPB, gy, gz);
    gemm_bf16_nt<<<grid, 256, 0, s>>>(A, lda, sAy, sAz, B, ldb, sBy, sBz,
                                      Cf, Cb, ldc, sCy, sCz, bias, M, N, K, alpha, relu);
}

extern "C" void kernel_launch(void* const* d_in, const int* in_sizes, int n_in,
                              void* d_out, int out_size, void* d_ws, size_t ws_size,
                              hipStream_t stream)
{
    (void)in_sizes; (void)n_in; (void)out_size; (void)ws_size;
    const int B = 8, L = 512, D = 512, H = 8, F = 2048, NL = 4;
    const int M = B * L;                       // 4096 rows of activations
    const long ATTN_L = (long)B * H * L * L;   // 16,777,216 per layer

    const float* src  = (const float*)d_in[0];
    const float* Wqkv = (const float*)d_in[1];
    const float* bqkv = (const float*)d_in[2];
    const float* Wo   = (const float*)d_in[3];
    const float* bo   = (const float*)d_in[4];
    const float* W1   = (const float*)d_in[5];
    const float* b1   = (const float*)d_in[6];
    const float* W2   = (const float*)d_in[7];
    const float* b2   = (const float*)d_in[8];
    const float* g1   = (const float*)d_in[9];
    const float* be1  = (const float*)d_in[10];
    const float* g2   = (const float*)d_in[11];
    const float* be2  = (const float*)d_in[12];

    float* out     = (float*)d_out;
    float* xf      = out;                       // running x lives in d_out[0:B*L*D]
    float* attnAll = out + (long)B * L * D;

    // ---- workspace carve (256B aligned) ----
    char* wp = (char*)d_ws;
    auto carve = [&](size_t bytes) { char* p = wp; wp += (bytes + 255) & ~(size_t)255; return p; };
    bf16* wqkv_bf = (bf16*)carve((size_t)NL * 3 * D * D * 2);   // 6.3 MB
    bf16* wo_bf   = (bf16*)carve((size_t)NL * D * D * 2);       // 2.1 MB
    bf16* w1_bf   = (bf16*)carve((size_t)NL * F * D * 2);       // 8.4 MB
    bf16* w2_bf   = (bf16*)carve((size_t)NL * D * F * 2);       // 8.4 MB
    bf16* x_bf    = (bf16*)carve((size_t)M * D * 2);            // 4.2 MB
    bf16* qkv_bf  = (bf16*)carve((size_t)M * 3 * D * 2);        // 12.6 MB
    bf16* vT_bf   = (bf16*)carve((size_t)B * H * 64 * L * 2);   // 4.2 MB
    bf16* attn_bf = (bf16*)carve((size_t)ATTN_L * 2);           // 33.6 MB
    bf16* ctx_bf  = (bf16*)carve((size_t)M * D * 2);            // 4.2 MB
    float* ao_f   = (float*)carve((size_t)M * D * 4);           // 8.4 MB
    bf16* h1_bf   = (bf16*)carve((size_t)M * F * 2);            // 16.8 MB
    float* ff_f   = (float*)carve((size_t)M * D * 4);           // 8.4 MB

    // ---- one-time conversions (weights fit in L2 afterwards) ----
    cvt_f32_bf16_k<<<2048, 256, 0, stream>>>(Wqkv, wqkv_bf, (long)NL * 3 * D * D);
    cvt_f32_bf16_k<<<1024, 256, 0, stream>>>(Wo,   wo_bf,   (long)NL * D * D);
    cvt_f32_bf16_k<<<2048, 256, 0, stream>>>(W1,   w1_bf,   (long)NL * F * D);
    cvt_f32_bf16_k<<<2048, 256, 0, stream>>>(W2,   w2_bf,   (long)NL * D * F);
    cvt_f32_bf16_k<<<2048, 256, 0, stream>>>(src,  x_bf,    (long)M * D);

    for (int l = 0; l < NL; ++l) {
        const bf16* wq = wqkv_bf + (size_t)l * 3 * D * D;
        const bf16* wo = wo_bf   + (size_t)l * D * D;
        const bf16* w1 = w1_bf   + (size_t)l * F * D;
        const bf16* w2 = w2_bf   + (size_t)l * D * F;
        float* attnL = attnAll + (size_t)l * ATTN_L;

        // qkv = x @ Wqkv^T + bqkv   -> bf16 [4096,1536]
        gemm_launch(stream, x_bf, D, 0, 0, wq, D, 0, 0,
                    nullptr, qkv_bf, 3 * D, 0, 0, bqkv + (size_t)l * 3 * D,
                    M, 3 * D, D, 1.0f, 0, 1, 1);

        // V^T for the context GEMM
        build_vT_k<<<(B * H * 64 * L) / 256, 256, 0, stream>>>(qkv_bf, vT_bf);

        // scores[b,h] = scale * Q @ K^T  -> fp32 straight into d_out attn slot
        gemm_launch(stream,
                    qkv_bf,        3 * D, 64, (long)L * 3 * D,   // Q: +h*64, +b*L*3D
                    qkv_bf + D,    3 * D, 64, (long)L * 3 * D,   // K block starts at col D
                    attnL, nullptr, L, (long)L * L, (long)H * L * L,
                    nullptr, L, L, 64, 0.125f, 0, H, B);

        // softmax rows in place, then bf16 shadow for attn @ V
        softmax512_k<<<B * H * L, 256, 0, stream>>>(attnL);
        cvt_f32_bf16_k<<<4096, 256, 0, stream>>>(attnL, attn_bf, ATTN_L);

        // ctx[b,h] = attn @ V  -> bf16 [B,L,D] layout (col offset h*64)
        gemm_launch(stream,
                    attn_bf, L, (long)L * L, (long)H * L * L,
                    vT_bf,   L, (long)64 * L, (long)H * 64 * L,
                    nullptr, ctx_bf, D, 64, (long)L * D,
                    nullptr, L, 64, L, 1.0f, 0, H, B);

        // attn_out = ctx @ Wo^T + bo -> fp32
        gemm_launch(stream, ctx_bf, D, 0, 0, wo, D, 0, 0,
                    ao_f, nullptr, D, 0, 0, bo + (size_t)l * D,
                    M, D, D, 1.0f, 0, 1, 1);

        // x = LN(x + attn_out)  (also emits bf16 x for FFN)
        const float* xin = (l == 0) ? src : xf;
        add_ln512_k<<<M, 256, 0, stream>>>(xin, ao_f, g1 + (size_t)l * D, be1 + (size_t)l * D, xf, x_bf);

        // h1 = relu(x @ W1^T + b1) -> bf16 [4096,2048]
        gemm_launch(stream, x_bf, D, 0, 0, w1, D, 0, 0,
                    nullptr, h1_bf, F, 0, 0, b1 + (size_t)l * F,
                    M, F, D, 1.0f, 1, 1, 1);

        // ff = h1 @ W2^T + b2 -> fp32
        gemm_launch(stream, h1_bf, F, 0, 0, w2, F, 0, 0,
                    ff_f, nullptr, D, 0, 0, b2 + (size_t)l * D,
                    M, D, F, 1.0f, 0, 1, 1);

        // x = LN(x + ff)
        add_ln512_k<<<M, 256, 0, stream>>>(xf, ff_f, g2 + (size_t)l * D, be2 + (size_t)l * D, xf, x_bf);
    }
}
